// AttentionCross_2061584302450
// MI455X (gfx1250) — matrix operand order repaired and checked by reading, not compiled, not run on hardware
//
#include <hip/hip_runtime.h>
#include <hip/hip_bf16.h>

// ---------------------------------------------------------------------------
// CDNA5 (gfx1250) bf16 WMMA cross-attention, with async global->LDS staging.
// ---------------------------------------------------------------------------

typedef __attribute__((ext_vector_type(16))) __bf16 bf16x16;
typedef __attribute__((ext_vector_type(8)))  __bf16 bf16x8;
typedef __attribute__((ext_vector_type(4)))  __bf16 bf16x4;
typedef __attribute__((ext_vector_type(8)))  float  f32x8;
typedef __attribute__((ext_vector_type(4)))  float  f32x4;

#define WMMA_BF16(a, b, c) \
  __builtin_amdgcn_wmma_f32_16x16x32_bf16(false, (a), false, (b), (short)0, (c), false, false)

// --- CDNA5 async global->LDS copy (tracked by ASYNCcnt) ---------------------
// LDS address is taken from the VDST VGPR; low 32 bits of a flat LDS pointer are the
// wave-relative LDS byte offset (aperture bits live in [63:32]).
__device__ __forceinline__ void async_ld_b128(void* lds_ptr, const void* gptr) {
  unsigned int loff = (unsigned int)(size_t)lds_ptr;
  asm volatile("global_load_async_to_lds_b128 %0, %1, off"
               :: "v"(loff), "v"(gptr) : "memory");
}
__device__ __forceinline__ void wait_async0() {
  asm volatile("s_wait_asynccnt 0x0" ::: "memory");
}

// --- Fragment loaders (layouts per CDNA5 ISA §7.12.2) ------------
// A (16x32 bf16): m=lane&15; elems 0..7 -> K=8*(lane>>4)+i, 8..15 -> K=16+...
__device__ __forceinline__ bf16x16 load_a_bf16(const __bf16* base, int stride, int lane) {
  const int m  = lane & 15;
  const int kh = (lane >> 4) << 3;
  const __bf16* row = base + m * stride;
  bf16x16 a;
#pragma unroll
  for (int i = 0; i < 8; ++i) a[i] = row[kh + i];
#pragma unroll
  for (int i = 0; i < 8; ++i) a[8 + i] = row[16 + kh + i];
  return a;
}

__device__ __forceinline__ bf16x16 load_a_f32(const float* base, int stride, int lane) {
  const int m  = lane & 15;
  const int kh = (lane >> 4) << 3;
  const float* row = base + m * stride;
  bf16x16 a;
#pragma unroll
  for (int i = 0; i < 8; ++i) a[i] = (__bf16)row[kh + i];
#pragma unroll
  for (int i = 0; i < 8; ++i) a[8 + i] = (__bf16)row[16 + kh + i];
  return a;
}

// B (32x16, B[k][n]) from a TRANSPOSED LDS image: row n holds column n over k.
__device__ __forceinline__ bf16x16 load_bcol_bf16(const __bf16* baseT, int stride, int lane) {
  const int n  = lane & 15;
  const int kh = (lane >> 4) << 3;
  const __bf16* row = baseT + n * stride;
  bf16x16 b;
#pragma unroll
  for (int i = 0; i < 8; ++i) b[i] = row[kh + i];
#pragma unroll
  for (int i = 0; i < 8; ++i) b[8 + i] = row[16 + kh + i];
  return b;
}

// C/D 16x16 f32: VGPR r -> row = r + 8*(lane>>4), col = lane&15.

// ---------------------------------------------------------------------------
// Projection GEMM: out = A[M,K](f32) @ W[K,N](f32) + bias, emitted as bf16.
// VIDEO: A element (row=b*2048+t, k) read from videofea[(b*K+k)*2048+t].
// OUTMODE 0: row-major [M,N]. 1: [b, N, 2048] (t-contig). 2: [b, N, 128].
// Block 256 thr (8 waves); 128x128 tile; wave w -> M-tile w, 8 N-tiles.
// ---------------------------------------------------------------------------
template <int VIDEO, int OUTMODE>
__global__ __launch_bounds__(256)
void proj_gemm_kernel(const float* __restrict__ Ain, const float* __restrict__ W,
                      const float* __restrict__ bias, __bf16* __restrict__ out,
                      int K, int N) {
  __shared__ __bf16 a_lds[128 * 36];   // [m][k], padded
  __shared__ __bf16 bT_lds[128 * 40];  // [n][k], padded

  const int mb   = blockIdx.x * 128;
  const int nb   = blockIdx.y * 128;
  const int tid  = threadIdx.x;
  const int wave = tid >> 5;
  const int lane = tid & 31;
  const int bb   = mb >> 11;      // video batch
  const int tb   = mb & 2047;     // video t-base

  f32x8 acc[8];
#pragma unroll
  for (int n = 0; n < 8; ++n)
#pragma unroll
    for (int r = 0; r < 8; ++r) acc[n][r] = 0.0f;

  for (int k0 = 0; k0 < K; k0 += 32) {
    f32x4 av[4], wv[4];
    // Batched vector loads (b128), waits coalesced by the compiler.
#pragma unroll
    for (int i = 0; i < 4; ++i) {
      const int e4 = i * 256 + tid;  // 1024 float4 slots per matrix
      if (VIDEO) {
        const int k = e4 >> 5, mq = (e4 & 31) << 2;      // contiguous along t
        av[i] = *(const f32x4*)(Ain + ((size_t)bb * K + k0 + k) * 2048 + tb + mq);
      } else {
        const int m = e4 >> 3, kq = (e4 & 7) << 2;       // contiguous along k
        av[i] = *(const f32x4*)(Ain + (size_t)(mb + m) * K + k0 + kq);
      }
      const int k2 = e4 >> 5, nq = (e4 & 31) << 2;       // contiguous along n
      wv[i] = *(const f32x4*)(W + (size_t)(k0 + k2) * N + nb + nq);
    }
    // Convert + stage to LDS.
#pragma unroll
    for (int i = 0; i < 4; ++i) {
      const int e4 = i * 256 + tid;
      if (VIDEO) {
        const int k = e4 >> 5, mq = (e4 & 31) << 2;
#pragma unroll
        for (int j = 0; j < 4; ++j) a_lds[(mq + j) * 36 + k] = (__bf16)av[i][j];
      } else {
        const int m = e4 >> 3, kq = (e4 & 7) << 2;
        bf16x4 v;
#pragma unroll
        for (int j = 0; j < 4; ++j) v[j] = (__bf16)av[i][j];
        *(bf16x4*)(a_lds + m * 36 + kq) = v;
      }
      const int k2 = e4 >> 5, nq = (e4 & 31) << 2;
#pragma unroll
      for (int j = 0; j < 4; ++j) bT_lds[(nq + j) * 40 + k2] = (__bf16)wv[i][j];
    }
    __syncthreads();

    const bf16x16 afrag = load_a_bf16(a_lds + wave * 16 * 36, 36, lane);
#pragma unroll
    for (int n = 0; n < 8; ++n) {
      const bf16x16 bfrag = load_bcol_bf16(bT_lds + n * 16 * 40, 40, lane);
      acc[n] = WMMA_BF16(afrag, bfrag, acc[n]);
    }
    __syncthreads();
  }

  const int row0g = mb + wave * 16 + ((lane >> 4) << 3);
  const int colb  = lane & 15;
#pragma unroll
  for (int n = 0; n < 8; ++n) {
    const int col = nb + n * 16 + colb;
    const float bv = bias[col];
    if (OUTMODE == 0) {
#pragma unroll
      for (int r = 0; r < 8; ++r)
        out[(size_t)(row0g + r) * N + col] = (__bf16)(acc[n][r] + bv);
    } else {
      bf16x8 v;
#pragma unroll
      for (int r = 0; r < 8; ++r) v[r] = (__bf16)(acc[n][r] + bv);
      __bf16* p = (OUTMODE == 1)
          ? out + ((size_t)(row0g >> 11) * N + col) * 2048 + (row0g & 2047)
          : out + ((size_t)(row0g >> 7)  * N + col) * 128  + (row0g & 127);
      *(bf16x8*)p = v;   // 16B-aligned contiguous store
    }
  }
}

// ---------------------------------------------------------------------------
// Fused dual-softmax attention. One WG per (b,g); T tiled by 64.
// All four operand tiles staged with global_load_async_to_lds_b128.
// ---------------------------------------------------------------------------
__global__ __launch_bounds__(256)
void attn_kernel(const __bf16* __restrict__ q_ws, const __bf16* __restrict__ k_ws,
                 const __bf16* __restrict__ vvT_ws, const __bf16* __restrict__ vtT_ws,
                 const int* __restrict__ mask, float* __restrict__ out_v,
                 float* __restrict__ out_t) {
  constexpr int L = 128, D = 128, T = 2048, AC = 1024, TT = 64;
  const int bg = blockIdx.x;
  const int b = bg >> 3;
  const int g = bg & 7;
  const int tid  = threadIdx.x;
  const int wave = tid >> 5;
  const int lane = tid & 31;

  extern __shared__ char smem_raw[];
  float*  s_lds   = (float*)smem_raw;               // [128][66] scores / exp
  float*  row_max = s_lds + 128 * 66;               // [128]
  float*  row_sum = row_max + 128;                  // [128]
  float*  corr    = row_sum + 128;                  // [128]
  __bf16* q_lds   = (__bf16*)(corr + 128);          // [128][136] (l,dd)
  __bf16* vvT_lds = q_lds + 128 * 136;              // [128][136] (dd,l)
  __bf16* k_lds   = vvT_lds + 128 * 136;            // [64][136]  (t,dd)
  __bf16* vtT_lds = k_lds + 64 * 136;               // [128][72]  (dd,t)
  __bf16* sT_lds  = (__bf16*)(vtT_lds + 128 * 72);  // [64][136]  (t,l)
  float*  red1    = (float*)(sT_lds + 64 * 136);    // [256]
  float*  red2    = red1 + 256;                     // [256]  total 160256 B

  // Stage q and vvT once (straight b128 async copies).
#pragma unroll
  for (int i = 0; i < 8; ++i) {
    const int e = i * 256 + tid;                    // 2048 chunks each
    const int r16 = e >> 4, c = (e & 15) << 3;
    async_ld_b128(q_lds + r16 * 136 + c,
                  q_ws + (size_t)(b * L + r16) * AC + g * D + c);
    async_ld_b128(vvT_lds + r16 * 136 + c,
                  vvT_ws + ((size_t)b * 1024 + g * D + r16) * 128 + c);
  }
  if (tid < L) { row_max[tid] = -1e30f; row_sum[tid] = 0.0f; }

  f32x8 accT[8];  // out_t accumulator: wave -> rows wave*16..+15, 8 dd-tiles
#pragma unroll
  for (int n = 0; n < 8; ++n)
#pragma unroll
    for (int r = 0; r < 8; ++r) accT[n][r] = 0.0f;

  wait_async0();
  __syncthreads();

  for (int t0 = 0; t0 < T; t0 += TT) {
    // ---- stage K tile and vtT tile (async b128) ----
#pragma unroll
    for (int i = 0; i < 4; ++i) {
      const int e = i * 256 + tid;                  // 1024 chunks each
      {
        const int t = e >> 4, c = (e & 15) << 3;
        async_ld_b128(k_lds + t * 136 + c,
                      k_ws + (size_t)(b * T + t0 + t) * AC + g * D + c);
      }
      {
        const int dd = e >> 3, c = (e & 7) << 3;
        async_ld_b128(vtT_lds + dd * 72 + c,
                      vtT_ws + ((size_t)b * 1024 + g * D + dd) * 2048 + t0 + c);
      }
    }
    wait_async0();
    __syncthreads();

    // ---- S[l][t] = q @ k^T : wave -> M-tile wave, N-tiles 0..3 ----
    {
      f32x8 c[4];
#pragma unroll
      for (int n = 0; n < 4; ++n)
#pragma unroll
        for (int r = 0; r < 8; ++r) c[n][r] = 0.0f;
#pragma unroll
      for (int k0 = 0; k0 < D; k0 += 32) {
        const bf16x16 afrag = load_a_bf16(q_lds + wave * 16 * 136 + k0, 136, lane);
#pragma unroll
        for (int n = 0; n < 4; ++n) {
          const bf16x16 bfrag = load_bcol_bf16(k_lds + n * 16 * 136 + k0, 136, lane);
          c[n] = WMMA_BF16(afrag, bfrag, c[n]);
        }
      }
      const int row0 = wave * 16 + ((lane >> 4) << 3);
      const int colb = lane & 15;
#pragma unroll
      for (int n = 0; n < 4; ++n)
#pragma unroll
        for (int r = 0; r < 8; ++r)
          s_lds[(row0 + r) * 66 + n * 16 + colb] = c[n][r];
    }
    __syncthreads();

    // ---- mask: att[l][t] = mask[b][t][l] ? att : -1e9 ----
#pragma unroll
    for (int i = 0; i < 32; ++i) {
      const int e = i * 256 + tid;
      const int t = e >> 7, l = e & 127;
      if (mask[(size_t)(b * T + t0 + t) * L + l] == 0)
        s_lds[l * 66 + t] = -1e9f;
    }
    __syncthreads();

    // ---- column softmax over L (tile-local), 4 threads per column ----
    {
      const int col = tid >> 2, l0 = (tid & 3) * 32;
      float mx = -1e30f;
      for (int l = l0; l < l0 + 32; ++l) mx = fmaxf(mx, s_lds[l * 66 + col]);
      red1[tid] = mx;
    }
    __syncthreads();
    {
      const int col = tid >> 2, l0 = (tid & 3) * 32;
      const float cm = fmaxf(fmaxf(red1[col * 4 + 0], red1[col * 4 + 1]),
                             fmaxf(red1[col * 4 + 2], red1[col * 4 + 3]));
      float sum = 0.0f;
      for (int l = l0; l < l0 + 32; ++l) {
        const float p = __expf(s_lds[l * 66 + col] - cm);
        sum += p;
        sT_lds[col * 136 + l] = (__bf16)p;
      }
      red2[tid] = sum;
    }
    __syncthreads();
    {
      // scale sT by 1/(colsum*32); simultaneously: row partial max -> red1
      const int col = tid >> 2, l0 = (tid & 3) * 32;
      const float cs = red2[col * 4 + 0] + red2[col * 4 + 1] +
                       red2[col * 4 + 2] + red2[col * 4 + 3];
      const float inv = 1.0f / (cs * 32.0f);
      for (int l = l0; l < l0 + 32; ++l)
        sT_lds[col * 136 + l] = (__bf16)((float)sT_lds[col * 136 + l] * inv);

      const int l = tid >> 1, t1 = (tid & 1) * 32;
      float pm = -1e30f;
      for (int t = t1; t < t1 + 32; ++t) pm = fmaxf(pm, s_lds[l * 66 + t]);
      red1[tid] = pm;
    }
    __syncthreads();

    // ---- online row softmax over T, 2 threads per row ----
    {
      const int l = tid >> 1, t1 = (tid & 1) * 32;
      const float nm = fmaxf(row_max[l], fmaxf(red1[l * 2], red1[l * 2 + 1]));
      float sum = 0.0f;
      for (int t = t1; t < t1 + 32; ++t) {
        const float p = __expf(s_lds[l * 66 + t] - nm);
        s_lds[l * 66 + t] = p;
        sum += p;
      }
      red2[tid] = sum;
    }
    __syncthreads();
    if (tid < L) {
      const int l = tid;
      const float mprev = row_max[l];
      const float nm = fmaxf(mprev, fmaxf(red1[l * 2], red1[l * 2 + 1]));
      const float cor = __expf(mprev - nm);
      row_sum[l] = row_sum[l] * cor + red2[l * 2] + red2[l * 2 + 1];
      row_max[l] = nm;
      corr[l] = cor;
    }
    __syncthreads();

    // ---- out_v tile: Dv[64x128] = Pv(sT) @ vv; write (b128 stores) ----
    {
      f32x8 c[4];
#pragma unroll
      for (int n = 0; n < 4; ++n)
#pragma unroll
        for (int r = 0; r < 8; ++r) c[n][r] = 0.0f;
      const int mt = wave >> 1;
      const int nbase = (wave & 1) * 4;
#pragma unroll
      for (int k0 = 0; k0 < L; k0 += 32) {
        const bf16x16 afrag = load_a_bf16(sT_lds + mt * 16 * 136 + k0, 136, lane);
#pragma unroll
        for (int n = 0; n < 4; ++n) {
          const bf16x16 bfrag =
              load_bcol_bf16(vvT_lds + (nbase + n) * 16 * 136 + k0, 136, lane);
          c[n] = WMMA_BF16(afrag, bfrag, c[n]);
        }
      }
      const int trow = t0 + mt * 16 + ((lane >> 4) << 3);
      const int colb = lane & 15;
#pragma unroll
      for (int n = 0; n < 4; ++n) {
        const int ch = g * D + (nbase + n) * 16 + colb;
        float* op = out_v + ((size_t)b * 1024 + ch) * 2048 + trow;
        f32x4 lo = {c[n][0], c[n][1], c[n][2], c[n][3]};
        f32x4 hi = {c[n][4], c[n][5], c[n][6], c[n][7]};
        *(f32x4*)op = lo;
        *(f32x4*)(op + 4) = hi;
      }
    }

    // ---- out_t: rescale accumulator, then accT += P @ vt ----
    {
      const int row0 = wave * 16 + ((lane >> 4) << 3);
#pragma unroll
      for (int n = 0; n < 8; ++n)
#pragma unroll
        for (int r = 0; r < 8; ++r) accT[n][r] *= corr[row0 + r];
#pragma unroll
      for (int k0 = 0; k0 < TT; k0 += 32) {
        const bf16x16 afrag = load_a_f32(s_lds + wave * 16 * 66 + k0, 66, lane);
#pragma unroll
        for (int n = 0; n < 8; ++n) {
          const bf16x16 bfrag = load_bcol_bf16(vtT_lds + n * 16 * 72 + k0, 72, lane);
          accT[n] = WMMA_BF16(afrag, bfrag, accT[n]);
        }
      }
    }
    __syncthreads();
  }

  // ---- final out_t = accT / (row_sum * 32) ----
  {
    const int row0 = wave * 16 + ((lane >> 4) << 3);
    const int colb = lane & 15;
#pragma unroll
    for (int n = 0; n < 8; ++n)
#pragma unroll
      for (int r = 0; r < 8; ++r) {
        const int l = row0 + r;
        const float val = accT[n][r] / (row_sum[l] * 32.0f);
        out_t[(size_t)(b * L + l) * 1024 + g * D + n * 16 + colb] = val;
      }
  }
}

// ---------------------------------------------------------------------------
// Host launcher.
//  in: 0 videofea f32[32,1024,2048] 1 textfea f32[32,128,768] 2 mask i32[32,2048,128]
//      3 Wq 4 bq 5 Wk 6 bk 7 Wvv 8 bvv 9 Wvt 10 bvt
//  out: out_v f32[32,1024,2048] ++ out_t f32[32,128,1024]
//  ws (bf16): q[b*L,A] 8MB | vvT[b,A,L] 8MB | k[b*T,A] 128MB | vtT[b,A,T] 128MB
// ---------------------------------------------------------------------------
extern "C" void kernel_launch(void* const* d_in, const int* in_sizes, int n_in,
                              void* d_out, int out_size, void* d_ws, size_t ws_size,
                              hipStream_t stream) {
  (void)in_sizes; (void)n_in; (void)out_size; (void)ws_size;
  const float* videofea = (const float*)d_in[0];
  const float* textfea  = (const float*)d_in[1];
  const int*   mask     = (const int*)d_in[2];
  const float* Wq  = (const float*)d_in[3];
  const float* bq  = (const float*)d_in[4];
  const float* Wk  = (const float*)d_in[5];
  const float* bk  = (const float*)d_in[6];
  const float* Wvv = (const float*)d_in[7];
  const float* bvv = (const float*)d_in[8];
  const float* Wvt = (const float*)d_in[9];
  const float* bvt = (const float*)d_in[10];

  char* ws = (char*)d_ws;
  const size_t QBYTES = (size_t)32 * 128 * 1024 * 2;   // 8 MB
  const size_t KBYTES = (size_t)32 * 2048 * 1024 * 2;  // 128 MB
  __bf16* q_ws   = (__bf16*)ws;
  __bf16* vvT_ws = (__bf16*)(ws + QBYTES);
  __bf16* k_ws   = (__bf16*)(ws + 2 * QBYTES);
  __bf16* vtT_ws = (__bf16*)(ws + 2 * QBYTES + KBYTES);

  dim3 blk(256);
  // Text projections: M=4096, K=768, N=1024.
  proj_gemm_kernel<0, 0><<<dim3(32, 8), blk, 0, stream>>>(textfea, Wq, bq, q_ws, 768, 1024);
  proj_gemm_kernel<0, 2><<<dim3(32, 8), blk, 0, stream>>>(textfea, Wvv, bvv, vvT_ws, 768, 1024);
  // Video projections: M=65536, K=1024, N=1024 (A read transposed from [b,vd,T]).
  proj_gemm_kernel<1, 0><<<dim3(512, 8), blk, 0, stream>>>(videofea, Wk, bk, k_ws, 1024, 1024);
  proj_gemm_kernel<1, 1><<<dim3(512, 8), blk, 0, stream>>>(videofea, Wvt, bvt, vtT_ws, 1024, 1024);

  float* out_v = (float*)d_out;
  float* out_t = out_v + (size_t)32 * 1024 * 2048;

  const size_t smem = 160256;  // dynamic LDS (~156.5 KB)
  attn_kernel<<<dim3(32 * 8), blk, smem, stream>>>(q_ws, k_ws, vvT_ws, vtT_ws, mask,
                                                   out_v, out_t);
}
